// Implicit_61125974557354
// MI455X (gfx1250) — matrix operand order repaired and checked by reading, not compiled, not run on hardware
//
#include <hip/hip_runtime.h>

// ---------------------------------------------------------------------------
// CDNA5 (gfx1250) implicit-depth SSM fixed-point solver.
//  - GEMMs on v_wmma_f32_16x16x32_bf16 (f32 accumulate)
//  - weight tiles staged global->LDS with the Tensor Data Mover (TDM),
//    double buffered; B fragments consumed via ds_load_b128 with a one-deep
//    software pipeline; each B fragment feeds 2 WMMAs (32-row wave tile)
//  - sequence recurrence parallelized with a 3-phase chunked scan
// ---------------------------------------------------------------------------

typedef __attribute__((ext_vector_type(16))) __bf16        v16bf;
typedef __attribute__((ext_vector_type(8)))  float         v8f;
typedef __attribute__((ext_vector_type(4)))  unsigned int  v4u;
typedef __attribute__((ext_vector_type(8)))  int           v8i;
typedef __attribute__((ext_vector_type(4)))  int           v4i;

constexpr int SEQ    = 8192;
constexpr int DM     = 1024;
constexpr int DS     = 64;
constexpr int DI     = 128;
constexpr int DZX    = DI + DM;        // 1152
constexpr int DZHX   = DI + DS + DM;   // 1216
constexpr int NITER  = 50;
constexpr int CHUNK  = 128;
constexpr int NCHUNK = SEQ / CHUNK;    // 64

constexpr int      KSTAGE      = 2;                      // k-tiles per LDS stage
constexpr unsigned STAGE_ELTS  = KSTAGE * 8 * 512;       // bf16 per stage (8192)
constexpr unsigned STAGE_BYTES = STAGE_ELTS * 2;         // 16 KB
constexpr unsigned LDS_BYTES   = 2 * STAGE_BYTES;        // double buffer

__device__ __forceinline__ __bf16 f2bf(float f) {
  // round-to-nearest-even f32 -> bf16
  unsigned int u = __builtin_bit_cast(unsigned int, f);
  unsigned int r = (u + 0x7FFFu + ((u >> 16) & 1u)) >> 16;
  return __builtin_bit_cast(__bf16, (unsigned short)r);
}

// ---- Tensor Data Mover: contiguous 1D block, global -> LDS ----------------
// D# per CDNA5 ISA: group0 {count=1, lds_addr, global_addr,
// type=2}; group1 {mask=0, data_size=2 (4B), tensor_dim0 = tile_dim0 = nDwords}.
__device__ __forceinline__ void tdm_load_1d(unsigned ldsByteOff,
                                            const void* gsrc,
                                            unsigned nDwords) {
  unsigned long long ga = (unsigned long long)(size_t)gsrc;
  v4u g0;
  g0.x = 1u;                                        // count=1, user mode
  g0.y = ldsByteOff;                                // lds_addr
  g0.z = (unsigned)(ga & 0xFFFFFFFFu);              // global_addr[31:0]
  g0.w = (unsigned)((ga >> 32) & 0x01FFFFFFu)       // global_addr[56:32]
         | 0x80000000u;                             // type=2 ("image")
  v8i g1;
  g1[0] = (int)0x00020000u;                         // wg_mask=0, data_size=4B
  g1[1] = (int)((nDwords & 0xFFFFu) << 16);         // tensor_dim0[15:0]
  g1[2] = (int)((nDwords >> 16) & 0xFFFFu);         // tensor_dim0[31:16]
  g1[3] = (int)((nDwords & 0xFFFFu) << 16);         // tile_dim0
  g1[4] = 0;                                        // tile_dim1/2 unused
  g1[5] = (int)nDwords;                             // tensor_dim0_stride lo
  g1[6] = 0;
  g1[7] = 0;
  v4i z4 = {0, 0, 0, 0};
#if defined(__clang_major__) && __clang_major__ >= 23
  v8i z8 = {0, 0, 0, 0, 0, 0, 0, 0};
  __builtin_amdgcn_tensor_load_to_lds(g0, g1, z4, z4, z8, 0);
#else
  __builtin_amdgcn_tensor_load_to_lds(g0, g1, z4, z4, 0);
#endif
}

// ---- WMMA fragment loaders -------------------------------------------------
// A (16x32 bf16, row-major source, ld = LDA elements):
//   lane 0-15  : row=lane,    VGPR g holds K pair {2g,2g+1} (g<4) / {16+2(g-4),..}
//   lane 16-31 : row=lane-16, same pattern shifted by +8 in K
__device__ __forceinline__ v16bf loadA(const __bf16* A, int lda, int rowBase,
                                       int kt, int lane) {
  const int row  = rowBase + (lane & 15);
  const int half = lane >> 4;
  const unsigned int* ar =
      (const unsigned int*)(A + (size_t)row * lda + (size_t)kt * 32);
  union { unsigned int u[8]; v16bf v; } f;
#pragma unroll
  for (int g = 0; g < 8; ++g) {
    const int idx = (g < 4 ? g : g + 4) + (half ? 4 : 0);
    f.u[g] = ar[idx];
  }
  return f.v;
}

// B (32x16 bf16) pre-swizzled kt-major: tile (kt,nt) = 512 contiguous bf16,
// lane l owns [l*16 .. l*16+15] -> two b128 loads.
__device__ __forceinline__ v16bf loadB(const __bf16* Bsw, int NT, int nt,
                                       int kt, int lane) {
  const uint4* bp =
      (const uint4*)(Bsw + (((size_t)kt * NT + nt) << 9)) + lane * 2;
  union { uint4 q[2]; v16bf v; } f;
  f.q[0] = bp[0];
  f.q[1] = bp[1];
  return f.v;
}

__device__ __forceinline__ v16bf loadB_lds(const __bf16* Bs, int ktl, int nt,
                                           int lane) {
  const uint4* bp = (const uint4*)(Bs + (((ktl * 8 + nt) << 9))) + lane * 2;
  union { uint4 q[2]; v16bf v; } f;
  f.q[0] = bp[0];
  f.q[1] = bp[1];
  return f.v;
}

// ---- pipelined GEMM core ---------------------------------------------------
// Block = 4 waves; wave tile = 32 rows x 64 cols:
//   rowBase = blk*64 + (wave>>1)*32, ntBase = (wave&1)*4.
// c[0..3] accumulate rows rowBase..+15, c[4..7] rows rowBase+16..+31.
// B staged to LDS by the TDM (double buffered); one-deep B reg pipeline.
template <int KT, int LDA>
__device__ __forceinline__ void gemm_core(const __bf16* Ag, const __bf16* Bsw,
                                          __bf16* Bs, int rowBase, int ntBase,
                                          int wave, int lane, v8f* c) {
  constexpr int NSTAGE = KT / KSTAGE;
  if (wave == 0) {                               // prologue: DMA stage 0
    tdm_load_1d(0u, Bsw, STAGE_BYTES / 4);
    __builtin_amdgcn_s_wait_tensorcnt((short)0);
  }
  __syncthreads();
  for (int s = 0; s < NSTAGE; ++s) {
    const int buf = s & 1;
    if (wave == 0 && (s + 1) < NSTAGE)           // DMA next stage (other buf)
      tdm_load_1d((unsigned)(buf ^ 1) * STAGE_BYTES,
                  Bsw + (size_t)(s + 1) * STAGE_ELTS, STAGE_BYTES / 4);
    const __bf16* Bstage = Bs + (size_t)buf * STAGE_ELTS;
#pragma unroll
    for (int ktl = 0; ktl < KSTAGE; ++ktl) {
      const int kt = s * KSTAGE + ktl;
      v16bf a0 = loadA(Ag, LDA, rowBase, kt, lane);
      v16bf a1 = loadA(Ag, LDA, rowBase + 16, kt, lane);
      v16bf b = loadB_lds(Bstage, ktl, ntBase, lane);
#pragma unroll
      for (int i = 0; i < 4; ++i) {
        v16bf bn = b;
        if (i < 3) bn = loadB_lds(Bstage, ktl, ntBase + i + 1, lane);
        c[i] = __builtin_amdgcn_wmma_f32_16x16x32_bf16(
            false, a0, false, b, (short)0, c[i], false, false);
        c[4 + i] = __builtin_amdgcn_wmma_f32_16x16x32_bf16(
            false, a1, false, b, (short)0, c[4 + i], false, false);
        b = bn;
      }
    }
    if (wave == 0 && (s + 1) < NSTAGE)
      __builtin_amdgcn_s_wait_tensorcnt((short)0);
    __syncthreads();   // next stage ready + this buf free before overwrite
  }
}

// ---- prep kernels ----------------------------------------------------------
// Swizzle W [N,K] row-major f32 (B element (k,n) = W[n,k]) into kt-major
// fragment order: tile(kt,nt) contiguous. Columns < split from w0 else w1.
__global__ void swizzleB(const float* w0, const float* w1, int split, int K,
                         int N, __bf16* dst) {
  int idx = blockIdx.x * blockDim.x + threadIdx.x;
  if (idx >= K * N) return;
  const int NT   = N / 16;
  const int e    = idx & 15;
  const int lane = (idx >> 4) & 31;
  const int tile = idx >> 9;
  const int nt   = tile % NT;
  const int kt   = tile / NT;
  const int k    = kt * 32 + (e < 8 ? e : e + 8) + ((lane >= 16) ? 8 : 0);
  const int n    = nt * 16 + (lane & 15);
  float v = (n < split) ? w0[(size_t)n * K + k]
                        : w1[(size_t)(n - split) * K + k];
  dst[idx] = f2bf(v);
}

// z := 0 in zx/zhx, h := 0 in zhx, x copied (bf16) into both.
__global__ void initAct(const float* x, __bf16* zx, __bf16* zhx) {
  int idx = blockIdx.x * blockDim.x + threadIdx.x;
  if (idx >= SEQ * DZHX) return;
  const int t = idx / DZHX;
  const int j = idx % DZHX;
  const __bf16 zero = __builtin_bit_cast(__bf16, (unsigned short)0);
  if (j < DI) {
    zhx[(size_t)t * DZHX + j] = zero;
    zx[(size_t)t * DZX + j]   = zero;
  } else if (j < DI + DS) {
    zhx[(size_t)t * DZHX + j] = zero;
  } else {
    const __bf16 b = f2bf(x[(size_t)t * DM + (j - DI - DS)]);
    zhx[(size_t)t * DZHX + j]                = b;
    zx[(size_t)t * DZX + DI + (j - DI - DS)] = b;
  }
}

// ---- GEMM kernels ----------------------------------------------------------
// [SEQ,1152] x [1152,128] -> sigmoid(lam+bias) | u+bias
// Waves with ntBase==0 emit lam columns (0..63), ntBase==4 emit u columns.
__global__ __launch_bounds__(128) void gemm_lamu(const __bf16* zx,
                                                 const __bf16* Bsw,
                                                 const float* lam_b,
                                                 const float* u_b,
                                                 float* lamBuf, float* uBuf) {
  extern __shared__ __align__(16) char smem[];
  __bf16* Bs = (__bf16*)smem;
  const int lane    = threadIdx.x & 31;
  const int wave    = threadIdx.x >> 5;
  const int rowBase = blockIdx.x * 64 + (wave >> 1) * 32;
  const int ntBase  = (wave & 1) * 4;
  v8f c[8] = {};
  gemm_core<DZX / 32, DZX>(zx, Bsw, Bs, rowBase, ntBase, wave, lane, c);
  const int n0 = lane & 15, half = lane >> 4;
#pragma unroll
  for (int rt = 0; rt < 2; ++rt) {
#pragma unroll
    for (int i = 0; i < 4; ++i) {
      const int col = i * 16 + n0; // column within the 64-wide half
#pragma unroll
      for (int j = 0; j < 8; ++j) {
        const int row = rowBase + rt * 16 + j + half * 8;
        const float v = c[rt * 4 + i][j];
        if (ntBase == 0) {
          const float t = v + lam_b[col];
          lamBuf[(size_t)row * DS + col] = 1.f / (1.f + __expf(-t));
        } else {
          uBuf[(size_t)row * DS + col] = v + u_b[col];
        }
      }
    }
  }
}

// [SEQ,1216] x [1216,128] -> z = silu(.+b); z written bf16 into zx & zhx
__global__ __launch_bounds__(128) void gemm_f(const __bf16* zhx,
                                              const __bf16* Bsw,
                                              const float* f_b, __bf16* zx,
                                              __bf16* zhx_w) {
  extern __shared__ __align__(16) char smem[];
  __bf16* Bs = (__bf16*)smem;
  const int lane    = threadIdx.x & 31;
  const int wave    = threadIdx.x >> 5;
  const int rowBase = blockIdx.x * 64 + (wave >> 1) * 32;
  const int ntBase  = (wave & 1) * 4;
  v8f c[8] = {};
  gemm_core<DZHX / 32, DZHX>(zhx, Bsw, Bs, rowBase, ntBase, wave, lane, c);
  const int n0 = lane & 15, half = lane >> 4;
#pragma unroll
  for (int rt = 0; rt < 2; ++rt) {
#pragma unroll
    for (int i = 0; i < 4; ++i) {
      const int col = (ntBase + i) * 16 + n0;
#pragma unroll
      for (int j = 0; j < 8; ++j) {
        const int row = rowBase + rt * 16 + j + half * 8;
        float v = c[rt * 4 + i][j] + f_b[col];
        float s = v / (1.f + __expf(-v)); // silu
        const __bf16 b = f2bf(s);
        zx[(size_t)row * DZX + col]     = b;
        zhx_w[(size_t)row * DZHX + col] = b;
      }
    }
  }
}

// z[SEQ,128] (cols 0..127 of zhx) x [128,1024] + out_b -> d_out (f32)
__global__ __launch_bounds__(128) void gemm_out(const __bf16* zhx,
                                                const __bf16* Bsw,
                                                const float* out_b,
                                                float* out) {
  const int lane    = threadIdx.x & 31;
  const int wave    = threadIdx.x >> 5;
  const int rowBase = blockIdx.x * 64 + wave * 16;
  const int ntBase  = blockIdx.y * 4;
  constexpr int KT = DI / 32;    // 4
  constexpr int NT = DM / 16;    // 64
  v8f c[4] = {};
  for (int kt = 0; kt < KT; ++kt) {
    v16bf a = loadA(zhx, DZHX, rowBase, kt, lane);
#pragma unroll
    for (int i = 0; i < 4; ++i) {
      v16bf b = loadB(Bsw, NT, ntBase + i, kt, lane);
      c[i] = __builtin_amdgcn_wmma_f32_16x16x32_bf16(
          false, a, false, b, (short)0, c[i], false, false);
    }
  }
  const int n0 = lane & 15, half = lane >> 4;
#pragma unroll
  for (int i = 0; i < 4; ++i) {
    const int col = (ntBase + i) * 16 + n0;
#pragma unroll
    for (int j = 0; j < 8; ++j) {
      const int row = rowBase + j + half * 8;
      out[(size_t)row * DM + col] = c[i][j] + out_b[col];
    }
  }
}

// ---- chunked scan: h_t = lam_t*h_{t-1} + u_t, h fed to token t is pre-t ----
__global__ void scan_agg(const float* lam, const float* u, float* aggL,
                         float* aggU) {
  const int c = blockIdx.x, d = threadIdx.x; // 64 chunks x 64 channels
  const float* lp = lam + (size_t)c * CHUNK * DS + d;
  const float* up = u + (size_t)c * CHUNK * DS + d;
  float L = 1.f, U = 0.f;
#pragma unroll 4
  for (int t = 0; t < CHUNK; ++t) {
    const float l = lp[t * DS], uu = up[t * DS];
    U = l * U + uu;
    L = l * L;
  }
  aggL[c * DS + d] = L;
  aggU[c * DS + d] = U;
}

__global__ void scan_chunks(const float* aggL, const float* aggU, float* pre) {
  const int d = threadIdx.x;
  float H = 0.f;
  for (int c = 0; c < NCHUNK; ++c) {
    pre[c * DS + d] = H;
    H = aggL[c * DS + d] * H + aggU[c * DS + d];
  }
}

// writes the shifted h (state before token t) as bf16 into zhx h-columns
__global__ void scan_apply(const float* lam, const float* u, const float* pre,
                           __bf16* zhx) {
  const int c = blockIdx.x, d = threadIdx.x;
  const float* lp = lam + (size_t)c * CHUNK * DS + d;
  const float* up = u + (size_t)c * CHUNK * DS + d;
  float H = pre[c * DS + d];
  const size_t t0 = (size_t)c * CHUNK;
#pragma unroll 4
  for (int t = 0; t < CHUNK; ++t) {
    zhx[(t0 + t) * DZHX + DI + d] = f2bf(H);
    H = lp[t * DS] * H + up[t * DS];
  }
}

// ---------------------------------------------------------------------------
extern "C" void kernel_launch(void* const* d_in, const int* in_sizes, int n_in,
                              void* d_out, int out_size, void* d_ws,
                              size_t ws_size, hipStream_t stream) {
  const float* x     = (const float*)d_in[0];
  const float* f_w   = (const float*)d_in[1];
  const float* f_b   = (const float*)d_in[2];
  const float* lam_w = (const float*)d_in[3];
  const float* lam_b = (const float*)d_in[4];
  const float* u_w   = (const float*)d_in[5];
  const float* u_b   = (const float*)d_in[6];
  const float* out_w = (const float*)d_in[7];
  const float* out_b = (const float*)d_in[8];
  float* out = (float*)d_out;

  // workspace layout (256 B aligned slices)
  size_t off = 0;
  auto take = [&](size_t bytes) -> void* {
    void* p = (char*)d_ws + off;
    off += (bytes + 255) & ~(size_t)255;
    return p;
  };
  __bf16* zx  = (__bf16*)take((size_t)SEQ * DZX * 2);   // [z|x] bf16
  __bf16* zhx = (__bf16*)take((size_t)SEQ * DZHX * 2);  // [z|h|x] bf16
  float* lamB = (float*)take((size_t)SEQ * DS * 4);
  float* uB   = (float*)take((size_t)SEQ * DS * 4);
  float* aggL = (float*)take((size_t)NCHUNK * DS * 4);
  float* aggU = (float*)take((size_t)NCHUNK * DS * 4);
  float* pre  = (float*)take((size_t)NCHUNK * DS * 4);
  __bf16* B_A = (__bf16*)take((size_t)DZX * DI * 2);    // [1152,128] frags
  __bf16* B_C = (__bf16*)take((size_t)DZHX * DI * 2);   // [1216,128] frags
  __bf16* B_D = (__bf16*)take((size_t)DI * DM * 2);     // [128,1024] frags
  if (off > ws_size) return; // workspace too small; nothing safe to do

  // one-time per call: weight swizzle + activation init (idempotent)
  swizzleB<<<(DZX * DI + 255) / 256, 256, 0, stream>>>(lam_w, u_w, DS, DZX,
                                                       DI, B_A);
  swizzleB<<<(DZHX * DI + 255) / 256, 256, 0, stream>>>(f_w, f_w, DI, DZHX,
                                                        DI, B_C);
  swizzleB<<<(DI * DM + 255) / 256, 256, 0, stream>>>(out_w, out_w, DM, DI,
                                                      DM, B_D);
  initAct<<<(SEQ * DZHX + 255) / 256, 256, 0, stream>>>(x, zx, zhx);

  // 50 fixed-point iterations: GEMM -> chunked scan -> GEMM
  for (int it = 0; it < NITER; ++it) {
    gemm_lamu<<<SEQ / 64, 128, LDS_BYTES, stream>>>(zx, B_A, lam_b, u_b, lamB,
                                                    uB);
    scan_agg<<<NCHUNK, DS, 0, stream>>>(lamB, uB, aggL, aggU);
    scan_chunks<<<1, DS, 0, stream>>>(aggL, aggU, pre);
    scan_apply<<<NCHUNK, DS, 0, stream>>>(lamB, uB, pre, zhx);
    gemm_f<<<SEQ / 64, 128, LDS_BYTES, stream>>>(zhx, B_C, f_b, zx, zhx);
  }

  // final projection to d_out
  dim3 gD(SEQ / 64, DM / 64);
  gemm_out<<<gD, 128, 0, stream>>>(zhx, B_D, out_b, out);
}
